// Codebook_42056319762523
// MI455X (gfx1250) — matrix operand order repaired and checked by reading, not compiled, not run on hardware
//
#include <hip/hip_runtime.h>

// ---------------------------------------------------------------------------
// VQ argmin:  out[b,hw] = argmin_k || x[b,:,hw] - c[k,:] ||^2
//           = argmax_k ( x.c_k - 0.5*||c_k||^2 )
// GEMM M=65536, N=512, K=512 as bf16x3 (hi/lo split) WMMA, fused argmax.
// v2: 64-pixel workgroup tile, full-C x panel resident in LDS (one barrier),
//     2 B-tiles x 4 A-tiles per wave (halves centers/L2 traffic), 2 N-passes.
// ---------------------------------------------------------------------------

typedef __attribute__((ext_vector_type(16))) __bf16 v16bf;
typedef __attribute__((ext_vector_type(8)))  float  v8f;

union V16U { v16bf v; uint4 q[2]; };

#define NB   16
#define NC   512               // channels (GEMM K)
#define NK   512               // clusters (GEMM N)
#define HWT  4096              // H*W
#define MT   64                // pixels per workgroup (one (b,h) row)
#define CHK  32                // channels per WMMA chunk
#define NCHUNK (NC / CHK)      // 16
#define RS   520               // LDS row stride in bf16 (1040B: 16B-aligned, conflict-free)

#define SM_XH   0
#define SM_XL   (MT * RS * 2)                  // 66560
#define SM_CSQ  (SM_XL + MT * RS * 2)          // 133120
#define SM_RV   (SM_CSQ + NK * 4)              // 135168
#define SM_RI   (SM_RV + 4 * 4 * 16 * 4)       // 136192
#define SM_SIZE (SM_RI + 4 * 4 * 16 * 4)       // 137216

// --- prep 1: split centers into bf16 hi/lo, swizzled to WMMA A-operand layout.
// 16-bit A 16x32 packing used here (same k order as B): lanes 0-15 hold K 0..15, lanes 16-31 hold
// K 16..31; 2 bf16 per VGPR, K ascending in lane-local memory order.
__global__ __launch_bounds__(256)
void vq_prep_centers(const float* __restrict__ cc,
                     unsigned short* __restrict__ ahi_u,
                     unsigned short* __restrict__ alo_u) {
    int t = blockIdx.x * 256 + threadIdx.x;        // 0 .. 262143
    int k = t >> 9;
    int c = t & (NC - 1);
    float v  = cc[t];
    __bf16 h = (__bf16)v;
    __bf16 l = (__bf16)(v - (float)h);
    int ntile = k >> 4, m = k & 15;
    int chunk = c >> 5, coff = c & 31;
    int g    = coff >> 3;
    int lane = m + ((coff >> 4) << 4);
    int pos  = coff & 15;
    int idx  = (((ntile * NCHUNK + chunk) * 32) + lane) * 16 + pos;
    union { __bf16 b; unsigned short u; } ch, cl2;
    ch.b = h; cl2.b = l;
    ahi_u[idx] = ch.u;
    alo_u[idx] = cl2.u;
}

// --- prep 2: 0.5 * ||c_k||^2
__global__ __launch_bounds__(256)
void vq_prep_csq(const float* __restrict__ cc, float* __restrict__ csqh) {
    int k = blockIdx.x * 256 + threadIdx.x;
    if (k < NK) {
        float s = 0.f;
        for (int c = 0; c < NC; ++c) {
            float v = cc[k * NC + c];
            s += v * v;
        }
        csqh[k] = 0.5f * s;
    }
}

// --- main fused GEMM + argmax kernel
__global__ __launch_bounds__(256)
void vq_argmin(const float* __restrict__ x,
               const unsigned short* __restrict__ ahi_u,
               const unsigned short* __restrict__ alo_u,
               const float* __restrict__ csqh,
               int* __restrict__ out) {
    extern __shared__ __align__(16) unsigned char smem[];
    __bf16* xh   = (__bf16*)(smem + SM_XH);
    __bf16* xl   = (__bf16*)(smem + SM_XL);
    float*  scsq = (float*)(smem + SM_CSQ);
    float*  rv   = (float*)(smem + SM_RV);   // [group(4)][wn(4)][16]
    int*    ri   = (int*)(smem + SM_RI);

    const int tid  = threadIdx.x;
    const int lane = tid & 31;
    const int w    = tid >> 5;
    const int wm   = w >> 2;          // 0..1 : 32-pixel half
    const int wn   = w & 3;           // 0..3 : 64-cluster group within pass

    const int pixbase = blockIdx.x * MT;
    const int b  = pixbase >> 12;
    const int hw = pixbase & (HWT - 1);
    const float* xb = x + (size_t)b * NC * HWT + hw;

    // ---- stage entire 64-pixel x 512-channel panel: f32 -> bf16 hi/lo, [pix][c]
    for (int i = tid; i < 4096; i += 256) {
        int c0  = (i >> 4) << 1;          // even channel
        int pix = (i & 15) << 2;          // 4-pixel group
        const float* g0 = xb + (size_t)c0 * HWT + pix;
        float4 a  = *(const float4*)g0;
        float4 bq = *(const float4*)(g0 + HWT);
        float av[4] = {a.x, a.y, a.z, a.w};
        float bv[4] = {bq.x, bq.y, bq.z, bq.w};
        #pragma unroll
        for (int j = 0; j < 4; ++j) {
            __bf16 h0 = (__bf16)av[j]; __bf16 l0 = (__bf16)(av[j] - (float)h0);
            __bf16 h1 = (__bf16)bv[j]; __bf16 l1 = (__bf16)(bv[j] - (float)h1);
            int idx = (pix + j) * RS + c0;                 // even -> 4B aligned
            union { __bf16 b2[2]; unsigned int u; } ph, pl;
            ph.b2[0] = h0; ph.b2[1] = h1;
            pl.b2[0] = l0; pl.b2[1] = l1;
            *(unsigned int*)(xh + idx) = ph.u;
            *(unsigned int*)(xl + idx) = pl.u;
        }
    }
    for (int i = tid; i < NK; i += 256) scsq[i] = csqh[i];
    __syncthreads();

    const __bf16* ahi = (const __bf16*)ahi_u;
    const __bf16* alo = (const __bf16*)alo_u;

    const int prow  = wm * 32 + (lane & 15);
    const int bcoff = (lane & 16);
    const int rbase = (lane >> 4) << 3;

    float best[2] = {-3.402823e38f, -3.402823e38f};
    int   bidx[2] = {0, 0};

    for (int pass = 0; pass < 2; ++pass) {
        v8f acc[8];                       // [bt*4 + at]
        #pragma unroll
        for (int t = 0; t < 8; ++t) {
            #pragma unroll
            for (int r = 0; r < 8; ++r) acc[t][r] = 0.f;
        }

        #pragma unroll 2
        for (int chunk = 0; chunk < NCHUNK; ++chunk) {
            // B operands for the wave's two 16-pixel tiles (LDS, conflict-free)
            V16U bh[2], bl[2];
            #pragma unroll
            for (int bt = 0; bt < 2; ++bt) {
                const __bf16* pBh = xh + (size_t)(prow + bt * 16) * RS + chunk * CHK + bcoff;
                bh[bt].q[0] = *(const uint4*)pBh;
                bh[bt].q[1] = *(const uint4*)(pBh + 8);
                const __bf16* pBl = xl + (size_t)(prow + bt * 16) * RS + chunk * CHK + bcoff;
                bl[bt].q[0] = *(const uint4*)pBl;
                bl[bt].q[1] = *(const uint4*)(pBl + 8);
            }
            #pragma unroll
            for (int at = 0; at < 4; ++at) {
                int ntile = pass * 16 + wn * 4 + at;
                size_t aoff = ((size_t)(ntile * NCHUNK + chunk) * 32 + lane) * 16;
                V16U ah, al;
                const uint4* pah = (const uint4*)(ahi + aoff);
                ah.q[0] = pah[0]; ah.q[1] = pah[1];
                const uint4* pal = (const uint4*)(alo + aoff);
                al.q[0] = pal[0]; al.q[1] = pal[1];
                #pragma unroll
                for (int bt = 0; bt < 2; ++bt) {
                    v8f a0 = acc[bt * 4 + at];
                    a0 = __builtin_amdgcn_wmma_f32_16x16x32_bf16(
                        false, ah.v, false, bl[bt].v, (short)0, a0, false, false);
                    a0 = __builtin_amdgcn_wmma_f32_16x16x32_bf16(
                        false, al.v, false, bh[bt].v, (short)0, a0, false, false);
                    a0 = __builtin_amdgcn_wmma_f32_16x16x32_bf16(
                        false, ah.v, false, bh[bt].v, (short)0, a0, false, false);
                    acc[bt * 4 + at] = a0;
                }
            }
        }

        // fold this pass's 64 clusters into the running argmax (k ascending ->
        // strict > keeps lowest index, matching jnp.argmin tie-break)
        #pragma unroll
        for (int bt = 0; bt < 2; ++bt) {
            #pragma unroll
            for (int at = 0; at < 4; ++at) {
                #pragma unroll
                for (int r = 0; r < 8; ++r) {
                    int k   = pass * 256 + wn * 64 + at * 16 + rbase + r;
                    float v = acc[bt * 4 + at][r] - scsq[k];
                    if (v > best[bt]) { best[bt] = v; bidx[bt] = k; }
                }
            }
        }
    }

    // merge lane halves (same pixel, complementary cluster rows), then waves
    #pragma unroll
    for (int bt = 0; bt < 2; ++bt) {
        float ov = __shfl_xor(best[bt], 16, 32);
        int   oi = __shfl_xor(bidx[bt], 16, 32);
        if (ov > best[bt] || (ov == best[bt] && oi < bidx[bt])) {
            best[bt] = ov; bidx[bt] = oi;
        }
        if ((lane & 16) == 0) {
            int grp = wm * 2 + bt;                       // 16-pixel group 0..3
            rv[(grp * 4 + wn) * 16 + lane] = best[bt];
            ri[(grp * 4 + wn) * 16 + lane] = bidx[bt];
        }
    }
    __syncthreads();

    if (tid < MT) {
        int pm = tid >> 4, p = tid & 15;
        float bv = rv[(pm * 4 + 0) * 16 + p];
        int   bi = ri[(pm * 4 + 0) * 16 + p];
        #pragma unroll
        for (int q = 1; q < 4; ++q) {
            float v  = rv[(pm * 4 + q) * 16 + p];
            int   i2 = ri[(pm * 4 + q) * 16 + p];
            if (v > bv || (v == bv && i2 < bi)) { bv = v; bi = i2; }
        }
        out[pixbase + tid] = bi;
    }
}

extern "C" void kernel_launch(void* const* d_in, const int* in_sizes, int n_in,
                              void* d_out, int out_size, void* d_ws, size_t ws_size,
                              hipStream_t stream) {
    const float* x  = (const float*)d_in[0];           // (16,512,64,64) f32
    const float* cc = (const float*)d_in[1];           // (1,512,512,1,1) f32
    int* out = (int*)d_out;                            // (16,1,64,64) argmin indices

    char* ws = (char*)d_ws;
    unsigned short* ahi = (unsigned short*)(ws);                  // 512 KiB
    unsigned short* alo = (unsigned short*)(ws + 512 * 1024);     // 512 KiB
    float*          csq = (float*)(ws + 1024 * 1024);             // 2 KiB

    vq_prep_centers<<<(NK * NC) / 256, 256, 0, stream>>>(cc, ahi, alo);
    vq_prep_csq<<<2, 256, 0, stream>>>(cc, csq);
    vq_argmin<<<(NB * HWT) / MT, 256, SM_SIZE, stream>>>(x, ahi, alo, csq, out);
}
